// AS_Mamba_Block_8718783611362
// MI455X (gfx1250) — matrix operand order repaired and checked by reading, not compiled, not run on hardware
//
#include <hip/hip_runtime.h>
#include <hip/hip_bf16.h>

typedef __bf16 bf16;
typedef __attribute__((ext_vector_type(16))) __bf16 v16bf;
typedef __attribute__((ext_vector_type(8)))  __bf16 v8bf;
typedef __attribute__((ext_vector_type(8)))  float  v8f;

#define G_GROUPS 5
#define DEPTHN   2
#define CH       256
#define DGEOM    64
#define HW       256   // B*H*W = 1*16*16

#if __has_builtin(__builtin_amdgcn_sched_barrier)
#define SCHED_FENCE() __builtin_amdgcn_sched_barrier(0)
#else
#define SCHED_FENCE()
#endif

__device__ __forceinline__ float bf2f(bf16 x) { return (float)x; }
__device__ __forceinline__ bf16  f2bf(float x) { return (bf16)x; }

// ---------------- feat (C,H,W) f32 -> fbt[(y*16+x)*C + c] f32 ----------------
__global__ void feat_transpose_kernel(const float* __restrict__ feat,
                                      float* __restrict__ fbt) {
  int tid = blockIdx.x * blockDim.x + threadIdx.x;  // 65536
  int q = tid & 255;
  int c = tid >> 8;
  fbt[(size_t)q * CH + c] = feat[(size_t)c * HW + q];
}

// ---------------- f32 weight [K][N] -> bf16 transposed [N][K] ----------------
__global__ void wprep_kernel(const float* __restrict__ src, bf16* __restrict__ dst,
                             int K, int N) {
  int tid = blockIdx.x * blockDim.x + threadIdx.x;
  if (tid >= K * N) return;
  int k = tid % K;
  int n = tid / K;
  dst[tid] = f2bf(src[(size_t)k * N + n]);   // dst[n*K + k]
}

// ---------------- bilinear window gather -> bf16 tokens ----------------
__global__ void gather_kernel(const float* __restrict__ fbt,
                              const float* __restrict__ flow,
                              bf16* __restrict__ win, int s, int L) {
  const int p = blockIdx.x;      // window position 0..L-1
  const int q = blockIdx.y;      // query pixel 0..255
  const int lane = threadIdx.x;  // 0..31, 8 channels each
  const float yq = (float)(q >> 4);
  const float xq = (float)(q & 15);
  const float cy = fminf(fmaxf(yq + flow[q * 4 + 1], 0.f), 15.f);
  const float cx = fminf(fmaxf(xq + flow[q * 4 + 0], 0.f), 15.f);
  const int half = s >> 1;
  const float ys = cy + (float)(p / s - half);
  const float xs = cx + (float)(p % s - half);
  const float y0 = floorf(ys), x0 = floorf(xs);
  const float wy = ys - y0, wx = xs - x0;
  const int y0i = (int)y0, x0i = (int)x0;

  float w[4]; const float* tp[4];
  const int   yy[2]  = { y0i, y0i + 1 };
  const int   xx[2]  = { x0i, x0i + 1 };
  const float wyv[2] = { 1.f - wy, wy };
  const float wxv[2] = { 1.f - wx, wx };
#pragma unroll
  for (int a = 0; a < 2; ++a)
#pragma unroll
    for (int b = 0; b < 2; ++b) {
      const int yi = yy[a], xi = xx[b];
      const bool valid = (yi >= 0) && (yi < 16) && (xi >= 0) && (xi < 16);
      const int yc = min(max(yi, 0), 15), xc = min(max(xi, 0), 15);
      w[a * 2 + b]  = wyv[a] * wxv[b] * (valid ? 1.f : 0.f);
      tp[a * 2 + b] = fbt + (size_t)(yc * 16 + xc) * CH;
    }
  const int c = lane * 8;
  v8bf o;
#pragma unroll
  for (int j = 0; j < 8; ++j) {
    const float v = w[0] * tp[0][c + j] + w[1] * tp[1][c + j] +
                    w[2] * tp[2][c + j] + w[3] * tp[3][c + j];
    o[j] = f2bf(v);
  }
  *(v8bf*)(win + ((size_t)q * L + p) * CH + c) = o;
}

// ---------------- RMSNorm * gamma (wave32 per token) ----------------
__global__ __launch_bounds__(256)
void rmsnorm_kernel(const bf16* __restrict__ x, const float* __restrict__ gamma,
                    bf16* __restrict__ xn, int M) {
  const int lane = threadIdx.x & 31;
  const int token = blockIdx.x * 8 + (threadIdx.x >> 5);
  if (token >= M) return;
  const size_t base = (size_t)token * CH + lane * 8;
  const v8bf v = *(const v8bf*)(x + base);
  float f[8]; float ss = 0.f;
#pragma unroll
  for (int i = 0; i < 8; ++i) { f[i] = bf2f(v[i]); ss += f[i] * f[i]; }
#pragma unroll
  for (int off = 16; off > 0; off >>= 1) ss += __shfl_xor(ss, off, 32);
  const float r = rsqrtf(ss * (1.f / 256.f) + 1e-6f);
  v8bf o;
#pragma unroll
  for (int i = 0; i < 8; ++i) o[i] = f2bf(f[i] * r * gamma[lane * 8 + i]);
  *(v8bf*)(xn + base) = o;
}

// ---------------- sequential scan + SiLU gate ----------------
__global__ __launch_bounds__(256)
void scan_silu_kernel(const bf16* __restrict__ uz, const float* __restrict__ alog,
                      bf16* __restrict__ y, int L) {
  const int tid = blockIdx.x * blockDim.x + threadIdx.x;  // 65536
  const int c = tid & 255;
  const int q = tid >> 8;
  const float a = 1.f / (1.f + __expf(-alog[c]));
  const bf16* up = uz + (size_t)q * L * (2 * CH) + c;
  bf16* yp = y + (size_t)q * L * CH + c;
  float h = 0.f;
  for (int t = 0; t < L; ++t) {
    const float u = bf2f(up[0]);
    const float z = bf2f(up[CH]);
    h = a * h + u;
    const float sz = z / (1.f + __expf(-z));
    yp[0] = f2bf(h * sz);
    up += 2 * CH;
    yp += CH;
  }
}

// ---------------- WMMA bf16 GEMM: C[MxN] = A[MxK] * Bt[NxK]^T ----------------
// Block = 4 waves (128 thr); each wave owns a 32x64 tile (2 M-tiles x 4 N-tiles,
// 8 accumulators) -> block tile 128x64. B tile staged through double-buffered
// LDS (shared by the 4 waves). Pipelining: A fragments + B stage chunk are
// register double-buffered across k-steps (global latency); B fragments are
// double-buffered across N-tiles, with sched_barrier pinning each t+1 LDS read
// ahead of the t WMMA pair so LDS latency overlaps the matrix pipe.
enum GemmMode { OUT_F32 = 0, OUT_BF16 = 1, OUT_BF16_RES = 2, OUT_F32_T_BIAS = 3 };

__global__ __launch_bounds__(128, 1)
void gemm_bf16_wmma_kernel(const bf16* __restrict__ A, const bf16* __restrict__ Bt,
                           void* __restrict__ Cout, const bf16* __restrict__ Res,
                           const float* __restrict__ bias,
                           int M, int N, int K, int mode) {
  constexpr int BSTRIDE = 72;          // padded per-column k-stride (bf16 elems):
                                       // 144B = 36 dwords -> conflict-free, 16B-aligned
  __shared__ bf16 ldsB[2][64][BSTRIDE];

  const int tid  = threadIdx.x;
  const int lane = tid & 31;
  const int wave = tid >> 5;           // 0..3
  const int m0 = blockIdx.y * 128 + wave * 32;   // wave owns rows m0..m0+31
  const int n0 = blockIdx.x * 64;
  if (m0 >= M || n0 >= N) return;      // grids are exact; never taken
  const int r  = lane & 15;
  const int hi = lane >> 4;

  // staging role: thread (0..127) -> (column 0..63, 16-elem k-chunk 0..1)
  const int scol = tid >> 1;
  const int skq  = (tid & 1) * 16;
  const bf16* Bstage = Bt + (size_t)(n0 + scol) * K + skq;

  // ISA §7.12.2 A layout: lane holds row r; K chunks [k0+8*hi,+8), [k0+16+8*hi,+8)
  const bf16* Arow0 = A + (size_t)(m0 + r) * K + hi * 8;
  const bf16* Arow1 = Arow0 + (size_t)16 * K;

  union frag { v16bf v; v8bf h[2]; };
  // B fragment from LDS: lane holds column 16*t+r, K chunks [8*hi,+8), [16+8*hi,+8)
  auto lds_frag = [&](frag& f, int buf, int t) {
    const bf16* lp = &ldsB[buf][16 * t + r][hi * 8];
    f.h[0] = *(const v8bf*)lp;
    f.h[1] = *(const v8bf*)(lp + 16);
  };

  // ---- prologue: stage B chunk k=0, preload A fragments k=0 ----
  *(v8bf*)&ldsB[0][scol][skq]     = *(const v8bf*)(Bstage + 0);
  *(v8bf*)&ldsB[0][scol][skq + 8] = *(const v8bf*)(Bstage + 8);
  frag a_cur[2];
  a_cur[0].h[0] = *(const v8bf*)(Arow0 + 0);
  a_cur[0].h[1] = *(const v8bf*)(Arow0 + 16);
  a_cur[1].h[0] = *(const v8bf*)(Arow1 + 0);
  a_cur[1].h[1] = *(const v8bf*)(Arow1 + 16);
  __syncthreads();

  v8f acc[2][4];
#pragma unroll
  for (int mt = 0; mt < 2; ++mt)
#pragma unroll
    for (int t = 0; t < 4; ++t) acc[mt][t] = v8f{};

  for (int k0 = 0; k0 < K; k0 += 32) {
    const int buf = (k0 >> 5) & 1;
    const bool more = (k0 + 32) < K;

    // first B fragment in flight as early as possible
    frag b0, b1;
    lds_frag(b0, buf, 0);

    // issue next k-step's global loads before consuming current fragments
    v8bf gs0 = {}, gs1 = {};
    frag a_nxt[2];
    if (more) {
      gs0 = *(const v8bf*)(Bstage + k0 + 32);
      gs1 = *(const v8bf*)(Bstage + k0 + 40);
      a_nxt[0].h[0] = *(const v8bf*)(Arow0 + k0 + 32);
      a_nxt[0].h[1] = *(const v8bf*)(Arow0 + k0 + 48);
      a_nxt[1].h[0] = *(const v8bf*)(Arow1 + k0 + 32);
      a_nxt[1].h[1] = *(const v8bf*)(Arow1 + k0 + 48);
      __builtin_prefetch((const void*)(Arow0 + k0 + 96), 0, 1);
    }

    // 8 WMMAs per k-step; fragment t+1's LDS reads are pinned (sched_barrier)
    // ahead of fragment t's WMMA pair, so both fragment buffers stay live in
    // distinct registers and each wait tolerates the in-flight next loads.
#pragma unroll
    for (int t = 0; t < 4; ++t) {
      frag& cur = (t & 1) ? b1 : b0;
      frag& nxt = (t & 1) ? b0 : b1;
      if (t < 3) lds_frag(nxt, buf, t + 1);
      SCHED_FENCE();
      acc[0][t] = __builtin_amdgcn_wmma_f32_16x16x32_bf16(
          false, a_cur[0].v, false, cur.v, (short)0, acc[0][t], false, false);
      acc[1][t] = __builtin_amdgcn_wmma_f32_16x16x32_bf16(
          false, a_cur[1].v, false, cur.v, (short)0, acc[1][t], false, false);
      SCHED_FENCE();
    }

    if (more) {
      // previous iteration's barrier ordered all reads of buf^1 before this write
      *(v8bf*)&ldsB[buf ^ 1][scol][skq]     = gs0;
      *(v8bf*)&ldsB[buf ^ 1][scol][skq + 8] = gs1;
      a_cur[0] = a_nxt[0];
      a_cur[1] = a_nxt[1];
      __syncthreads();
    }
  }

  // D layout: VGPR v -> row v + 8*hi, col = lane&15 (+16*t)
#pragma unroll
  for (int mt = 0; mt < 2; ++mt) {
#pragma unroll
    for (int t = 0; t < 4; ++t) {
      const int nn = n0 + 16 * t + r;
      const float bv = (mode == OUT_F32_T_BIAS) ? bias[nn] : 0.f;
#pragma unroll
      for (int v = 0; v < 8; ++v) {
        const int mm = m0 + mt * 16 + v + hi * 8;
        const float val = acc[mt][t][v];
        if (mode == OUT_F32) {
          ((float*)Cout)[(size_t)mm * N + nn] = val;
        } else if (mode == OUT_BF16) {
          ((bf16*)Cout)[(size_t)mm * N + nn] = f2bf(val);
        } else if (mode == OUT_BF16_RES) {
          ((bf16*)Cout)[(size_t)mm * N + nn] =
              f2bf(val + bf2f(Res[(size_t)mm * N + nn]));
        } else {  // OUT_F32_T_BIAS: C[n*M + m] = val + bias[n]
          ((float*)Cout)[(size_t)nn * M + mm] = val + bv;
        }
      }
    }
  }
}

// ---------------- center pick * span mask -> concat buffers ----------------
__global__ __launch_bounds__(256)
void center_mask_kernel(const bf16* __restrict__ x, const bf16* __restrict__ geom,
                        const float* __restrict__ sx, const float* __restrict__ sy,
                        bf16* __restrict__ mcat, bf16* __restrict__ gcat,
                        int gi, int L, int Lc, float span) {
  const int tid = blockIdx.x * blockDim.x + threadIdx.x;  // 65536
  const int c = tid & 255;
  const int q = tid >> 8;
  const float avg = (sx[q] + sy[q]) * 0.5f;
  const float mask = (fabsf(avg - span) < 1.5f) ? 1.f : 0.f;
  const size_t center = (size_t)q * L + Lc;
  mcat[(size_t)q * (G_GROUPS * CH) + gi * CH + c] =
      f2bf(bf2f(x[center * CH + c]) * mask);
  if (c < DGEOM)
    gcat[(size_t)q * (G_GROUPS * DGEOM) + gi * DGEOM + c] =
        f2bf(bf2f(geom[center * DGEOM + c]) * mask);
}

// =============================== host side ===============================
extern "C" void kernel_launch(void* const* d_in, const int* in_sizes, int n_in,
                              void* d_out, int out_size, void* d_ws, size_t ws_size,
                              hipStream_t stream) {
  (void)in_sizes; (void)n_in; (void)out_size; (void)ws_size;
  const float* feat[2] = { (const float*)d_in[0], (const float*)d_in[1] };
  const float* flow[2] = { (const float*)d_in[4], (const float*)d_in[5] };
  const float* sxv[2]  = { (const float*)d_in[6], (const float*)d_in[8] };
  const float* syv[2]  = { (const float*)d_in[7], (const float*)d_in[9] };
  const float* Win   = (const float*)d_in[10];
  const float* alog  = (const float*)d_in[11];
  const float* Wout  = (const float*)d_in[12];
  const float* Wgeom = (const float*)d_in[13];
  const float* gamma = (const float*)d_in[14];
  const float* Wagg  = (const float*)d_in[15];
  const float* bagg  = (const float*)d_in[16];
  const float* WaggG = (const float*)d_in[17];
  const float* baggG = (const float*)d_in[18];
  float* out = (float*)d_out;

  static const int spans[G_GROUPS] = { 5, 7, 9, 11, 15 };
  const int Lmax = 225;
  const size_t Mmax = (size_t)HW * Lmax;

  char* base = (char*)d_ws;
  size_t off = 0;
  auto take = [&](size_t bytes) -> char* {
    char* p = base + off;
    off = (off + bytes + 255) & ~(size_t)255;
    return p;
  };
  float* fbt   = (float*)take((size_t)HW * CH * sizeof(float));
  bf16* WinT   = (bf16*)take((size_t)G_GROUPS * DEPTHN * 2 * CH * CH * sizeof(bf16));
  bf16* WoutT  = (bf16*)take((size_t)G_GROUPS * DEPTHN * CH * CH * sizeof(bf16));
  bf16* WgeomT = (bf16*)take((size_t)G_GROUPS * DEPTHN * DGEOM * CH * sizeof(bf16));
  bf16* WaggT  = (bf16*)take((size_t)CH * (G_GROUPS * CH) * sizeof(bf16));
  bf16* WaggGT = (bf16*)take((size_t)DGEOM * (G_GROUPS * DGEOM) * sizeof(bf16));
  bf16* xA  = (bf16*)take(Mmax * CH * sizeof(bf16));
  bf16* xB  = (bf16*)take(Mmax * CH * sizeof(bf16));
  bf16* xn  = (bf16*)take(Mmax * CH * sizeof(bf16));
  bf16* uz  = (bf16*)take(Mmax * 2 * CH * sizeof(bf16));
  bf16* yb  = (bf16*)take(Mmax * CH * sizeof(bf16));
  bf16* gb  = (bf16*)take(Mmax * DGEOM * sizeof(bf16));
  bf16* mcat = (bf16*)take((size_t)HW * G_GROUPS * CH * sizeof(bf16));
  bf16* gcat = (bf16*)take((size_t)HW * G_GROUPS * DGEOM * sizeof(bf16));

  // ---- weight prep (f32 -> bf16, transposed to [N][K]) ----
  for (int g = 0; g < G_GROUPS; ++g)
    for (int d = 0; d < DEPTHN; ++d) {
      const int idx = g * DEPTHN + d;
      { const int K = CH, N = 2 * CH, tot = K * N;
        wprep_kernel<<<(tot + 255) / 256, 256, 0, stream>>>(
            Win + (size_t)idx * K * N, WinT + (size_t)idx * (size_t)N * K, K, N); }
      { const int K = CH, N = CH, tot = K * N;
        wprep_kernel<<<(tot + 255) / 256, 256, 0, stream>>>(
            Wout + (size_t)idx * K * N, WoutT + (size_t)idx * (size_t)N * K, K, N); }
      { const int K = CH, N = DGEOM, tot = K * N;
        wprep_kernel<<<(tot + 255) / 256, 256, 0, stream>>>(
            Wgeom + (size_t)idx * K * N, WgeomT + (size_t)idx * (size_t)N * K, K, N); }
    }
  { const int K = G_GROUPS * CH, N = CH, tot = K * N;
    wprep_kernel<<<(tot + 255) / 256, 256, 0, stream>>>(Wagg, WaggT, K, N); }
  { const int K = G_GROUPS * DGEOM, N = DGEOM, tot = K * N;
    wprep_kernel<<<(tot + 255) / 256, 256, 0, stream>>>(WaggG, WaggGT, K, N); }

  // ---- two streams ----
  for (int s = 0; s < 2; ++s) {
    feat_transpose_kernel<<<(HW * CH) / 256, 256, 0, stream>>>(feat[s], fbt);
    for (int g = 0; g < G_GROUPS; ++g) {
      const int sp = spans[g];
      const int L = sp * sp;
      const int M = HW * L;           // multiple of 128
      const int Lc = L / 2;
      gather_kernel<<<dim3(L, HW), 32, 0, stream>>>(fbt, flow[s], xA, sp, L);
      bf16* x  = xA;
      bf16* xo = xB;
      for (int d = 0; d < DEPTHN; ++d) {
        const int widx = g * DEPTHN + d;
        rmsnorm_kernel<<<M / 8, 256, 0, stream>>>(
            x, gamma + (size_t)widx * CH, xn, M);
        { dim3 gU((2 * CH) / 64, M / 128);
          gemm_bf16_wmma_kernel<<<gU, 128, 0, stream>>>(
              xn, WinT + (size_t)widx * 2 * CH * CH, uz, nullptr, nullptr,
              M, 2 * CH, CH, OUT_BF16); }
        scan_silu_kernel<<<(HW * CH) / 256, 256, 0, stream>>>(
            uz, alog + (size_t)widx * CH, yb, L);
        { dim3 gO(CH / 64, M / 128);
          gemm_bf16_wmma_kernel<<<gO, 128, 0, stream>>>(
              yb, WoutT + (size_t)widx * CH * CH, xo, x, nullptr,
              M, CH, CH, OUT_BF16_RES); }
        if (d == DEPTHN - 1) {
          dim3 gG(DGEOM / 64, M / 128);
          gemm_bf16_wmma_kernel<<<gG, 128, 0, stream>>>(
              yb, WgeomT + (size_t)widx * DGEOM * CH, gb, nullptr, nullptr,
              M, DGEOM, CH, OUT_BF16);
        }
        bf16* t = x; x = xo; xo = t;
      }
      center_mask_kernel<<<(HW * CH) / 256, 256, 0, stream>>>(
          x, gb, sxv[s], syv[s], mcat, gcat, g, L, Lc, (float)sp);
    }
    // m = mcat[256x1280] @ Wagg + bagg  -> out[c][q]
    { dim3 gm(CH / 64, HW / 128);
      gemm_bf16_wmma_kernel<<<gm, 128, 0, stream>>>(
          mcat, WaggT, out + (size_t)s * CH * HW, nullptr, bagg,
          HW, CH, G_GROUPS * CH, OUT_F32_T_BIAS); }
    // g = gcat[256x320] @ Wagg_g + bagg_g -> out[d][q]
    { dim3 gg(DGEOM / 64, HW / 128);
      gemm_bf16_wmma_kernel<<<gg, 128, 0, stream>>>(
          gcat, WaggGT, out + 2 * (size_t)CH * HW + (size_t)s * DGEOM * HW,
          nullptr, baggG, HW, DGEOM, G_GROUPS * DGEOM, OUT_F32_T_BIAS); }
  }
}
